// STCNNT_Cell_89369679495484
// MI455X (gfx1250) — matrix operand order repaired and checked by reading, not compiled, not run on hardware
//
#include <hip/hip_runtime.h>
#include <hip/hip_bf16.h>
#include <math.h>

typedef __bf16 bf16;
typedef __attribute__((ext_vector_type(16))) __bf16 v16bf;
typedef __attribute__((ext_vector_type(8)))  __bf16 v8bf;
typedef __attribute__((ext_vector_type(8)))  float  v8f;
typedef __attribute__((ext_vector_type(4)))  unsigned int v4u;
typedef __attribute__((ext_vector_type(8)))  int v8i;
typedef __attribute__((ext_vector_type(4)))  int v4i;

#if defined(__AMDGCN__) && __has_builtin(__builtin_amdgcn_tensor_load_to_lds) && \
    __has_builtin(__builtin_amdgcn_s_wait_tensorcnt)
#define HAVE_TDM 1
#endif

// ---------------------------------------------------------------------------
// WMMA helper: D = A(16x32 bf16) * B(32x16 bf16) + C(16x16 f32)
// ---------------------------------------------------------------------------
__device__ __forceinline__ v8f wmma_bf16(v16bf a, v16bf b, v8f c) {
  return __builtin_amdgcn_wmma_f32_16x16x32_bf16(
      /*neg_a=*/false, a, /*neg_b=*/false, b,
      /*c_mod=*/(short)0, c, /*reuse_a=*/false, /*reuse_b=*/false);
}

__device__ __forceinline__ v8bf load8bf(const bf16* p) {
  return *reinterpret_cast<const v8bf*>(p);
}

__device__ __forceinline__ v16bf pack16(v8bf lo, v8bf hi) {
  v16bf r;
#pragma unroll
  for (int i = 0; i < 8; ++i) { r[i] = lo[i]; r[8 + i] = hi[i]; }
  return r;
}

// A fragment (16x32, contiguous k order used here):
//   lane L holds row M; e=0..15 -> K = khi+e,
//   khi = (L>=16) ? 16 : 0. rowbase points at (row 0, k0).
__device__ __forceinline__ v16bf loadA_rows(const bf16* rowbase, int rstride,
                                            int m, int lane) {
  int khi = (lane >= 16) ? 16 : 0;
  const bf16* p = rowbase + (size_t)m * rstride + khi;
  return pack16(load8bf(p), load8bf(p + 8));
}

// B fragment (32x16): lane L holds column N = L%16, e=0..15 -> K = khi+e,
//   khi = (L>=16) ? 16 : 0. colbase points at (col 0, k0).
__device__ __forceinline__ v16bf loadB_cols(const bf16* colbase, int cstride,
                                            int lane) {
  int n   = lane & 15;
  int khi = (lane >= 16) ? 16 : 0;
  const bf16* p = colbase + (size_t)n * cstride + khi;
  return pack16(load8bf(p), load8bf(p + 8));
}

__device__ __forceinline__ float new_gelu_f(float x) {
  const float c = 0.7978845608028654f; // sqrt(2/pi)
  return 0.5f * x * (1.f + tanhf(c * (x + 0.044715f * x * x * x)));
}

// ---------------------------------------------------------------------------
// TDM: DMA a 2D bf16 matrix [dim1 rows x dim0 cols] from global to LDS.
// D# per CDNA5 ISA ch.8: group0 = {flags, lds_addr, global_addr, type=2},
// group1 = {data_size=2B, tensor dims/tile dims/stride}, groups 2/3 unused.
// ---------------------------------------------------------------------------
#if defined(HAVE_TDM)
__device__ __forceinline__ void tdm_load_2d_bf16(const bf16* gsrc,
                                                 unsigned lds_byte_off,
                                                 unsigned dim0, unsigned dim1) {
  unsigned long long ga = (unsigned long long)(const void*)gsrc;
  v4u g0;
  g0[0] = 1u;                                        // count=1, user mode
  g0[1] = lds_byte_off;                              // lds_addr
  g0[2] = (unsigned)(ga & 0xFFFFFFFFu);              // global_addr[31:0]
  g0[3] = (unsigned)((ga >> 32) & 0x1FFFFFFu) | (2u << 30); // addr hi, type=2
  v8i g1;
  g1[0] = 0x10000;                                   // data_size=1 (2 bytes)
  g1[1] = (int)((dim0 & 0xFFFFu) << 16);             // tensor_dim0 lo16
  g1[2] = (int)((dim0 >> 16) | ((dim1 & 0xFFFFu) << 16)); // tdim0 hi, tdim1 lo
  g1[3] = (int)((dim1 >> 16) | ((dim0 & 0xFFFFu) << 16)); // tdim1 hi, tile_dim0
  g1[4] = (int)(dim1 & 0xFFFFu);                     // tile_dim1 (tile_dim2=0)
  g1[5] = (int)dim0;                                 // tensor_dim0_stride
  g1[6] = 0;
  g1[7] = 0;
  v4i z4 = {};
#if __clang_major__ >= 23
  v8i z8 = {};
  __builtin_amdgcn_tensor_load_to_lds(g0, g1, z4, z4, z8, 0);
#else
  __builtin_amdgcn_tensor_load_to_lds(g0, g1, z4, z4, 0);
#endif
}
#endif

// ---------------------------------------------------------------------------
// Weight convert + reorder: [C_OUT][C_IN][3][3] f32 -> [C_OUT][9][C_IN] bf16
// (GEMM K-order = (kh*3+kw)*C_IN + c)
// ---------------------------------------------------------------------------
__global__ void cvt_w_khwc(const float* __restrict__ win,
                           bf16* __restrict__ wout, int c_in, int n) {
  int i = blockIdx.x * 256 + threadIdx.x;
  if (i >= n) return;
  int o   = i / (c_in * 9);
  int rem = i - o * c_in * 9;
  int c   = rem / 9;
  int r   = rem - c * 9;
  wout[(o * 9 + r) * c_in + c] = (bf16)win[i];
}

// ---------------------------------------------------------------------------
// LayerNorm over [C,H,W] per (b,t) sample; reads NCHW f32, writes NHWC bf16.
// ---------------------------------------------------------------------------
__global__ __launch_bounds__(256)
void ln_chw_kernel(const float* __restrict__ x, const float* __restrict__ g,
                   const float* __restrict__ b, bf16* __restrict__ out) {
  constexpr int N = 32 * 64 * 64; // C*H*W
  __shared__ float s_sum[256];
  __shared__ float s_sq[256];
  const float* xs = x + (size_t)blockIdx.x * N;
  float sum = 0.f, sq = 0.f;
  for (int i = threadIdx.x; i < N; i += 256) {
    float v = xs[i];
    sum += v;
    sq += v * v;
  }
  s_sum[threadIdx.x] = sum;
  s_sq[threadIdx.x] = sq;
  __syncthreads();
  for (int s = 128; s > 0; s >>= 1) {
    if (threadIdx.x < s) {
      s_sum[threadIdx.x] += s_sum[threadIdx.x + s];
      s_sq[threadIdx.x]  += s_sq[threadIdx.x + s];
    }
    __syncthreads();
  }
  float mean = s_sum[0] * (1.f / N);
  float var  = s_sq[0] * (1.f / N) - mean * mean;
  float rstd = rsqrtf(var + 1e-5f);
  bf16* os = out + (size_t)blockIdx.x * N;
  for (int i = threadIdx.x; i < N; i += 256) {
    float v = (xs[i] - mean) * rstd * g[i] + b[i];
    int c = i >> 12, hw = i & 4095;       // i = c*4096 + hw
    os[hw * 32 + c] = (bf16)v;            // NHWC
  }
}

// ---------------------------------------------------------------------------
// 3x3 conv (pad 1), implicit GEMM, NHWC input, K-order (kh,kw)-major.
//   M = C_OUT, K = 9*C_IN, N = pixels. Block = 128 (4 waves).
//   Weights staged in LDS (TDM when available), A-fragments read from LDS.
//   Each wave owns NB 16-pixel N-tiles and all M-tiles.
//   Grid = (P/(64*NB), n_images).
// ---------------------------------------------------------------------------
template <int C_IN, int C_OUT, int H_IN, int W_IN, int STRIDE, int NB,
          bool GELU_ACT, bool OUT_NHWC>
__global__ __launch_bounds__(128)
void conv3x3_wmma(const bf16* __restrict__ xin,   // [Nimg, H*W, C_IN] NHWC
                  const bf16* __restrict__ wbf,   // [C_OUT, 9, C_IN]
                  const float* __restrict__ bias, // [C_OUT] or nullptr
                  bf16* __restrict__ obf,         // bf16 out or nullptr
                  float* __restrict__ of32,       // f32 NCHW out or nullptr
                  const float* __restrict__ resid,// f32 NCHW residual or null
                  int in_base, int out_base) {
  constexpr int H_OUT = H_IN / STRIDE;
  constexpr int W_OUT = W_IN / STRIDE;
  constexpr int P  = H_OUT * W_OUT;
  constexpr int KT = 9 * C_IN;
  constexpr int MT = C_OUT / 16;

  __shared__ bf16 wlds[C_OUT * KT];

  const int wave = threadIdx.x >> 5;
  const int lane = threadIdx.x & 31;

  // ---- stage weights into LDS (async tensor DMA when available) ----------
#if defined(HAVE_TDM)
  if (wave == 0) {
    tdm_load_2d_bf16(wbf, (unsigned)(size_t)(&wlds[0]), KT, C_OUT);
    __builtin_amdgcn_s_wait_tensorcnt(0);
  }
#else
  for (int i = threadIdx.x; i < (C_OUT * KT) / 8; i += 128) {
    *reinterpret_cast<v8bf*>(&wlds[i * 8]) = load8bf(wbf + i * 8);
  }
#endif
  __syncthreads();

  const int pix0 = (blockIdx.x * 4 + wave) * (16 * NB);
  if (pix0 >= P) return; // uniform per wave

  const int img_in  = in_base + blockIdx.y;
  const int img_out = out_base + blockIdx.y;
  const bf16* img = xin + (size_t)img_in * (H_IN * W_IN) * C_IN;

  // per-tile input origin for this lane's B column
  int ih0[NB], iw0[NB];
#pragma unroll
  for (int j = 0; j < NB; ++j) {
    int n  = pix0 + j * 16 + (lane & 15);
    int oh = n / W_OUT, ow = n - oh * W_OUT;
    ih0[j] = oh * STRIDE - 1;
    iw0[j] = ow * STRIDE - 1;
  }
  const int kbhi = (lane >= 16) ? 16 : 0;

  v8f acc[MT][NB] = {};

  for (int k0 = 0; k0 < KT; k0 += 32) {
    const int K0  = k0 + kbhi;            // start of this lane's 16-wide K-run
    const int r9  = K0 / C_IN;            // tap index (uniform over the run)
    const int cst = K0 - r9 * C_IN;       // channel start (multiple of 16)
    const int kh  = r9 / 3;
    const int kw  = r9 - kh * 3;

    v16bf bfrag[NB];
#pragma unroll
    for (int j = 0; j < NB; ++j) {
      int ih = ih0[j] + kh, iw = iw0[j] + kw;
      v8bf lo = {}, hi = {};
      if ((unsigned)ih < (unsigned)H_IN && (unsigned)iw < (unsigned)W_IN) {
        const bf16* p = img + ((size_t)(ih * W_IN + iw)) * C_IN + cst;
        lo = load8bf(p);
        hi = load8bf(p + 8);
      }
      bfrag[j] = pack16(lo, hi);
    }
#pragma unroll
    for (int mt = 0; mt < MT; ++mt) {
      v16bf a = loadA_rows(&wlds[k0], KT, mt * 16 + (lane & 15), lane);
#pragma unroll
      for (int j = 0; j < NB; ++j) acc[mt][j] = wmma_bf16(a, bfrag[j], acc[mt][j]);
    }
  }

  // D lane layout: VGPR r holds (M = r + (lane>=16?8:0), N = lane%16)
  const int mo = (lane >= 16) ? 8 : 0;
#pragma unroll
  for (int j = 0; j < NB; ++j) {
    int n = pix0 + j * 16 + (lane & 15);
#pragma unroll
    for (int mt = 0; mt < MT; ++mt) {
      if (OUT_NHWC && obf) {
        v8bf ov;
#pragma unroll
        for (int r = 0; r < 8; ++r) {
          int m = mt * 16 + mo + r;
          float v = acc[mt][j][r];
          if (bias) v += bias[m];
          if (GELU_ACT) v = new_gelu_f(v);
          ov[r] = (bf16)v;
        }
        *reinterpret_cast<v8bf*>(obf + ((size_t)img_out * P + n) * C_OUT +
                                 mt * 16 + mo) = ov;
      } else {
#pragma unroll
        for (int r = 0; r < 8; ++r) {
          int m = mt * 16 + mo + r;
          float v = acc[mt][j][r];
          if (bias) v += bias[m];
          if (GELU_ACT) v = new_gelu_f(v);
          size_t oidx = ((size_t)img_out * C_OUT + m) * P + n; // NCHW
          if (obf)  obf[oidx] = (bf16)v;
          if (of32) of32[oidx] = (resid ? resid[oidx] : 0.f) + v;
        }
      }
    }
  }
}

// ---------------------------------------------------------------------------
// Temporal attention per (b, head). T=32, d_qk = 4096, d_v = 16384.
// q/k: NCHW conv layout [b*T+t, c, 1024]; v/y: NHWC [b*T+t, 4096pix, 32c].
// ---------------------------------------------------------------------------
__global__ __launch_bounds__(128)
void attn_kernel(const bf16* __restrict__ qb, const bf16* __restrict__ kb,
                 const bf16* __restrict__ vb, bf16* __restrict__ yb) {
  const int bidx = blockIdx.x >> 3;
  const int head = blockIdx.x & 7;
  const int wave = threadIdx.x >> 5;
  const int lane = threadIdx.x & 31;

  __shared__ float att_f[32 * 32];
  __shared__ bf16  att_b[32 * 32];

  // ---- step 1: att = (q @ k^T) / 64 ---------------------------------------
  {
    const int mt = wave >> 1, nt = wave & 1;
    v8f acc = {};
    for (int k0 = 0; k0 < 4096; k0 += 32) {
      int ch = k0 >> 10, pk = k0 & 1023;
      size_t base = (((size_t)(bidx * 32) * 32) + head * 4 + ch) * 1024 + pk;
      v16bf a  = loadA_rows(qb + base, 32 * 1024, mt * 16 + (lane & 15), lane);
      v16bf bf = loadB_cols(kb + base + (size_t)(nt * 16) * 32768, 32 * 1024, lane);
      acc = wmma_bf16(a, bf, acc);
    }
    int nn = nt * 16 + (lane & 15);
    int mo = (lane >= 16) ? 8 : 0;
#pragma unroll
    for (int r = 0; r < 8; ++r) {
      int t = mt * 16 + r + mo;
      att_f[t * 32 + nn] = acc[r] * (1.f / 64.f);
    }
  }
  __syncthreads();

  // ---- step 2: softmax rows ----------------------------------------------
  if (threadIdx.x < 32) {
    int t = threadIdx.x;
    float mx = -1e30f;
    for (int s = 0; s < 32; ++s) mx = fmaxf(mx, att_f[t * 32 + s]);
    float sum = 0.f;
    for (int s = 0; s < 32; ++s) sum += __expf(att_f[t * 32 + s] - mx);
    float inv = 1.f / sum;
    for (int s = 0; s < 32; ++s)
      att_b[t * 32 + s] = (bf16)(__expf(att_f[t * 32 + s] - mx) * inv);
  }
  __syncthreads();

  // ---- step 3: y = att @ v (M=32 t, K=32 s, N=16384 dv) -------------------
  v16bf a0, a1;
  {
    int khi = (lane >= 16) ? 16 : 0;
    int m = lane & 15;
#pragma unroll
    for (int e = 0; e < 8; ++e) {
      a0[e]     = att_b[m * 32 + khi + e];
      a0[8 + e] = att_b[m * 32 + 8 + khi + e];
      a1[e]     = att_b[(16 + m) * 32 + khi + e];
      a1[8 + e] = att_b[(16 + m) * 32 + 8 + khi + e];
    }
  }
  const int kbhi = (lane >= 16) ? 16 : 0;
  for (int nt = wave; nt < 1024; nt += 4) {
    int dv = nt * 16 + (lane & 15);
    int ch = dv >> 12, pv = dv & 4095;           // d = ch*4096 + pix
    int cc = head * 4 + ch;
    v16bf bfrag;
#pragma unroll
    for (int e = 0; e < 16; ++e) {
      int s = kbhi + e;
      bfrag[e] = vb[(((size_t)(bidx * 32 + s)) * 4096 + pv) * 32 + cc]; // NHWC
    }
    v8f c0 = {}, c1 = {};
    c0 = wmma_bf16(a0, bfrag, c0);
    c1 = wmma_bf16(a1, bfrag, c1);
    int mo = (lane >= 16) ? 8 : 0;
#pragma unroll
    for (int r = 0; r < 8; ++r) {
      int t0 = r + mo, t1 = 16 + r + mo;
      yb[(((size_t)(bidx * 32 + t0)) * 4096 + pv) * 32 + cc] = (bf16)c0[r];
      yb[(((size_t)(bidx * 32 + t1)) * 4096 + pv) * 32 + cc] = (bf16)c1[r];
    }
  }
}

// ---------------------------------------------------------------------------
// Host launch
// ---------------------------------------------------------------------------
extern "C" void kernel_launch(void* const* d_in, const int* in_sizes, int n_in,
                              void* d_out, int out_size, void* d_ws,
                              size_t ws_size, hipStream_t stream) {
  const float* x   = (const float*)d_in[0];
  const float* n1g = (const float*)d_in[1];
  const float* n1b = (const float*)d_in[2];
  const float* n2g = (const float*)d_in[3];
  const float* n2b = (const float*)d_in[4];
  const float* Wk  = (const float*)d_in[5];
  const float* Wq  = (const float*)d_in[6];
  const float* Wv  = (const float*)d_in[7];
  const float* Wo  = (const float*)d_in[8];
  const float* bo  = (const float*)d_in[9];
  const float* Wm1 = (const float*)d_in[10];
  const float* bm1 = (const float*)d_in[11];
  const float* Wm2 = (const float*)d_in[12];
  const float* bm2 = (const float*)d_in[13];
  float* out = (float*)d_out;

  char* ws = (char*)d_ws;
  size_t off = 0;
  auto alloc = [&](size_t bytes) {
    void* p = ws + off;
    off = (off + bytes + 255) & ~(size_t)255;
    return p;
  };

  const size_t NI = 128; // B*T images
  bf16* xn   = (bf16*)alloc(NI * 4096 * 32 * 2);       // NHWC, xn / xn2
  bf16* qbuf = (bf16*)alloc(NI * 32 * 1024 * 2);       // NCHW
  bf16* kbuf = (bf16*)alloc(NI * 32 * 1024 * 2);       // NCHW
  bf16* vbuf = (bf16*)alloc(NI * 4096 * 32 * 2);       // NHWC
  bf16* ybuf = (bf16*)alloc(NI * 4096 * 32 * 2);       // NHWC
  bf16* hbuf = (bf16*)alloc((size_t)16 * 4096 * 128 * 2); // NHWC mixer chunk
  bf16* wkb  = (bf16*)alloc(32 * 288 * 2);
  bf16* wqb  = (bf16*)alloc(32 * 288 * 2);
  bf16* wvb  = (bf16*)alloc(32 * 288 * 2);
  bf16* wob  = (bf16*)alloc(32 * 288 * 2);
  bf16* wm1b = (bf16*)alloc(128 * 288 * 2);
  bf16* wm2b = (bf16*)alloc(32 * 1152 * 2);

  auto cvtw = [&](const float* src, bf16* dst, int c_in, int n) {
    cvt_w_khwc<<<(n + 255) / 256, 256, 0, stream>>>(src, dst, c_in, n);
  };
  cvtw(Wk, wkb, 32, 32 * 288);
  cvtw(Wq, wqb, 32, 32 * 288);
  cvtw(Wv, wvb, 32, 32 * 288);
  cvtw(Wo, wob, 32, 32 * 288);
  cvtw(Wm1, wm1b, 32, 128 * 288);
  cvtw(Wm2, wm2b, 128, 32 * 1152);

  // --- pre-norm 1 (NCHW f32 -> NHWC bf16) ---
  ln_chw_kernel<<<128, 256, 0, stream>>>(x, n1g, n1b, xn);

  // --- q, k (stride 2, NCHW out), v (stride 1, NHWC out) ---
  dim3 gqk(1024 / 128, 128); // NB=2
  conv3x3_wmma<32, 32, 64, 64, 2, 2, false, false><<<gqk, 128, 0, stream>>>(
      xn, wqb, nullptr, qbuf, nullptr, nullptr, 0, 0);
  conv3x3_wmma<32, 32, 64, 64, 2, 2, false, false><<<gqk, 128, 0, stream>>>(
      xn, wkb, nullptr, kbuf, nullptr, nullptr, 0, 0);
  dim3 gv(4096 / 128, 128); // NB=2
  conv3x3_wmma<32, 32, 64, 64, 1, 2, false, true><<<gv, 128, 0, stream>>>(
      xn, wvb, nullptr, vbuf, nullptr, nullptr, 0, 0);

  // --- temporal attention (B*heads = 32 blocks) ---
  attn_kernel<<<32, 128, 0, stream>>>(qbuf, kbuf, vbuf, ybuf);

  // --- output projection + residual: d_out = x + conv_o(y) + bo (NCHW f32) ---
  conv3x3_wmma<32, 32, 64, 64, 1, 2, false, false><<<gv, 128, 0, stream>>>(
      ybuf, wob, bo, nullptr, out, x, 0, 0);

  // --- pre-norm 2 (reads d_out), writes xn2 (NHWC) into xn buffer ---
  ln_chw_kernel<<<128, 256, 0, stream>>>(out, n2g, n2b, xn);

  // --- mixer: conv(32->128)+GELU (NHWC), conv(128->32)+bias+residual (NCHW) ---
  for (int c = 0; c < 8; ++c) {
    dim3 gm1(4096 / 64, 16);  // NB=1, MT=8
    conv3x3_wmma<32, 128, 64, 64, 1, 1, true, true><<<gm1, 128, 0, stream>>>(
        xn, wm1b, bm1, hbuf, nullptr, nullptr, c * 16, 0);
    dim3 gm2(4096 / 128, 16); // NB=2, MT=2
    conv3x3_wmma<128, 32, 64, 64, 1, 2, false, false><<<gm2, 128, 0, stream>>>(
        hbuf, wm2b, bm2, nullptr, out, out, 0, c * 16);
  }
}
